// OLNNMoE_73117523247426
// MI455X (gfx1250) — matrix operand order repaired and checked by reading, not compiled, not run on hardware
//
#include <hip/hip_runtime.h>
#include <math.h>

// ---------------- problem constants (from reference) ----------------
#define T_TOK 32768       // B*S = 8*4096
#define DIM   512         // D
#define HID   2048        // H
#define NE    8           // experts
#define TM    32          // token tile per FFN block

typedef __attribute__((ext_vector_type(16))) __bf16 v16bf;
typedef __attribute__((ext_vector_type(8)))  __bf16 v8bf;
typedef __attribute__((ext_vector_type(8)))  float  v8f;

// ---------------- helpers ----------------
__device__ __forceinline__ __bf16 f2bf(float f) {
    unsigned u = __float_as_uint(f);
    unsigned r = u + 0x7FFFu + ((u >> 16) & 1u);   // round-to-nearest-even
    unsigned short h = (unsigned short)(r >> 16);
    __bf16 out;
    __builtin_memcpy(&out, &h, sizeof(out));
    return out;
}

// branch-free tanh-approx gelu: tanh(t) = 1 - 2/(1+exp(2t))
__device__ __forceinline__ float gelu_tanh(float x) {
    float t = 0.7978845608f * (x + 0.044715f * x * x * x);
    float th = 1.0f - 2.0f / (1.0f + __expf(2.0f * t));
    return 0.5f * x * (1.0f + th);
}

// workspace header (lives at d_ws offset 0)
struct WsHeader {
    int   counts[NE];
    float acc_pa;         // sum over (t,e) of p*(1-p) at penalty temp
    float acc_z;          // sum over t of lse^2
    float acc_pm[NE];     // sum over t of p_e (penalty temp)
    float acc_imp[NE];    // importance = sum of routing weights
    float acc_cnt[NE];    // top-1 counts
    float acc_rp[NE];     // sum over t of router_probs (noisy softmax)
};

// ---------------- kernels ----------------
__global__ void zero_f32(float* p, int n) {
    int i = blockIdx.x * 256 + threadIdx.x;
    if (i < n) p[i] = 0.0f;
}

// one-time f32 -> bf16 weight conversion (8 elements / thread, b128 in / b128 out)
__global__ __launch_bounds__(256)
void cvt_bf16_kernel(const float* __restrict__ src, __bf16* __restrict__ dst, int n8) {
    int i = blockIdx.x * 256 + threadIdx.x;
    if (i >= n8) return;
    size_t base = (size_t)i * 8;
    float4 a = *(const float4*)(src + base);
    float4 b = *(const float4*)(src + base + 4);
    v8bf v;
    v[0] = f2bf(a.x); v[1] = f2bf(a.y); v[2] = f2bf(a.z); v[3] = f2bf(a.w);
    v[4] = f2bf(b.x); v[5] = f2bf(b.y); v[6] = f2bf(b.z); v[7] = f2bf(b.w);
    *(v8bf*)(dst + base) = v;
}

__global__ __launch_bounds__(256)
void router_kernel(const float* __restrict__ x,
                   const float* __restrict__ gate_w,
                   const float* __restrict__ noise_w,
                   const float* __restrict__ bias,
                   const float* __restrict__ noise,
                   WsHeader* __restrict__ hdr,
                   int* __restrict__ lists,
                   float* __restrict__ wlist) {
    __shared__ float s_pa, s_z;
    __shared__ float s_pm[NE], s_imp[NE], s_cnt[NE], s_rp[NE];
    const int tid = threadIdx.x;
    if (tid == 0) { s_pa = 0.0f; s_z = 0.0f; }
    if (tid < NE) { s_pm[tid] = 0.0f; s_imp[tid] = 0.0f; s_cnt[tid] = 0.0f; s_rp[tid] = 0.0f; }
    __syncthreads();

    const int wave = tid >> 5;
    const int lane = tid & 31;
    const int t = blockIdx.x * 8 + wave;   // one wave32 per token

    float acc[2 * NE];
#pragma unroll
    for (int i = 0; i < 2 * NE; ++i) acc[i] = 0.0f;

    const float* xr = x + (size_t)t * DIM;
    for (int d = lane; d < DIM; d += 32) {
        float xv = xr[d];
#pragma unroll
        for (int e = 0; e < NE; ++e) acc[e]      += xv * gate_w[e * DIM + d];
#pragma unroll
        for (int e = 0; e < NE; ++e) acc[NE + e] += xv * noise_w[e * DIM + d];
    }
#pragma unroll
    for (int i = 0; i < 2 * NE; ++i) {
        float v = acc[i];
#pragma unroll
        for (int off = 16; off > 0; off >>= 1) v += __shfl_xor(v, off, 32);
        acc[i] = v;
    }

    if (lane == 0) {
        float logits[NE];
#pragma unroll
        for (int e = 0; e < NE; ++e) logits[e] = acc[e];

        // penalty terms: softmax(logits / 1.66)
        float mx = -1e30f;
#pragma unroll
        for (int e = 0; e < NE; ++e) mx = fmaxf(mx, logits[e]);
        float pt[NE], spt = 0.0f;
#pragma unroll
        for (int e = 0; e < NE; ++e) { pt[e] = __expf((logits[e] - mx) * (1.0f / 1.66f)); spt += pt[e]; }
        float pa = 0.0f;
#pragma unroll
        for (int e = 0; e < NE; ++e) {
            float p = pt[e] / spt;
            pa += p * (1.0f - p);
            atomicAdd(&s_pm[e], p);
        }
        atomicAdd(&s_pa, pa);

        // z-loss: logsumexp of raw logits
        float se = 0.0f;
#pragma unroll
        for (int e = 0; e < NE; ++e) se += __expf(logits[e] - mx);
        float lse = mx + __logf(se);
        atomicAdd(&s_z, lse * lse);

        // noisy logits
        float g[NE];
#pragma unroll
        for (int e = 0; e < NE; ++e) {
            float ns = acc[NE + e];
            float sp = (ns > 20.0f) ? ns : log1pf(__expf(ns));   // softplus
            g[e] = logits[e] + noise[(size_t)t * NE + e] * sp + bias[e];
        }
        // top-2 (lowest index wins ties, like lax.top_k)
        int i0 = 0;
#pragma unroll
        for (int e = 1; e < NE; ++e) if (g[e] > g[i0]) i0 = e;
        int i1 = (i0 == 0) ? 1 : 0;
#pragma unroll
        for (int e = 0; e < NE; ++e) if (e != i0 && g[e] > g[i1]) i1 = e;
        float m2 = fmaxf(g[i0], g[i1]);
        float e0 = __expf(g[i0] - m2), e1 = __expf(g[i1] - m2);
        float w0 = e0 / (e0 + e1), w1 = e1 / (e0 + e1);
        atomicAdd(&s_imp[i0], w0);
        atomicAdd(&s_imp[i1], w1);
        atomicAdd(&s_cnt[i0], 1.0f);

        // router_probs = softmax(noisy logits)
        float mg = -1e30f;
#pragma unroll
        for (int e = 0; e < NE; ++e) mg = fmaxf(mg, g[e]);
        float sg = 0.0f, pg[NE];
#pragma unroll
        for (int e = 0; e < NE; ++e) { pg[e] = __expf(g[e] - mg); sg += pg[e]; }
#pragma unroll
        for (int e = 0; e < NE; ++e) atomicAdd(&s_rp[e], pg[e] / sg);

        // scatter token into the two expert lists
        int p0 = atomicAdd(&hdr->counts[i0], 1);
        lists[i0 * T_TOK + p0] = t;  wlist[i0 * T_TOK + p0] = w0;
        int p1 = atomicAdd(&hdr->counts[i1], 1);
        lists[i1 * T_TOK + p1] = t;  wlist[i1 * T_TOK + p1] = w1;
    }
    __syncthreads();
    if (tid == 0) { atomicAdd(&hdr->acc_pa, s_pa); atomicAdd(&hdr->acc_z, s_z); }
    if (tid < NE) {
        atomicAdd(&hdr->acc_pm[tid],  s_pm[tid]);
        atomicAdd(&hdr->acc_imp[tid], s_imp[tid]);
        atomicAdd(&hdr->acc_cnt[tid], s_cnt[tid]);
        atomicAdd(&hdr->acc_rp[tid],  s_rp[tid]);
    }
}

// Fused expert FFN over a 32-token tile: out += w * (gelu(x W1^T + b1) W2^T + b2)
// Weights pre-converted to bf16. LDS: x bf16 (32KB) + h bf16 (128KB) + meta.
__global__ __launch_bounds__(256)
void ffn_kernel(const float* __restrict__ x,
                const __bf16* __restrict__ w1, const float* __restrict__ b1,
                const __bf16* __restrict__ w2, const float* __restrict__ b2,
                const int* __restrict__ counts,
                const int* __restrict__ lists,
                const float* __restrict__ wlist,
                float* __restrict__ out) {
    const int e = blockIdx.y;
    const int cnt = counts[e];
    const int row0 = blockIdx.x * TM;
    if (row0 >= cnt) return;
    const int nrows = min(TM, cnt - row0);

    extern __shared__ char smem[];
    __bf16* sX = (__bf16*)smem;              // [TM][DIM]
    __bf16* sH = sX + TM * DIM;              // [TM][HID]
    float*  sW = (float*)(sH + TM * HID);    // [TM]
    int*    sT = (int*)(sW + TM);            // [TM]

    const int tid = threadIdx.x;
    if (tid < TM) {
        if (tid < nrows) {
            sT[tid] = lists[e * T_TOK + row0 + tid];
            sW[tid] = wlist[e * T_TOK + row0 + tid];
        } else { sT[tid] = 0; sW[tid] = 0.0f; }
    }
    __syncthreads();

    // gather x rows -> LDS (f32 -> bf16), float4 vectorized
    for (int i = tid; i < TM * DIM / 4; i += 256) {
        int r = (i * 4) / DIM, d = (i * 4) - r * DIM;
        float4 v = (r < nrows) ? *(const float4*)(x + (size_t)sT[r] * DIM + d)
                               : make_float4(0.f, 0.f, 0.f, 0.f);
        __bf16* dst = sX + r * DIM + d;
        dst[0] = f2bf(v.x); dst[1] = f2bf(v.y); dst[2] = f2bf(v.z); dst[3] = f2bf(v.w);
    }
    __syncthreads();

    const int wave = tid >> 5;
    const int lane = tid & 31;
    const int m    = lane & 15;      // A row / C column-lane
    const int half = lane >> 4;      // K half select (ISA 7.12.2 16-bit layouts)

    // ---- phase 1: h = gelu(x W1^T + b1), h in LDS bf16 ----
    const int rt = wave & 1;
    const __bf16* W1e = w1 + (size_t)e * HID * DIM;
    for (int it = 0; it < (HID / 16) / 4; ++it) {
        const int ct = (wave >> 1) + 4 * it;
        const int ncol = ct * 16 + m;
        v8f acc = {};
        const __bf16* brow_base = W1e + (size_t)ncol * DIM;
        for (int k0 = 0; k0 < DIM; k0 += 32) {
            // A fragment: two contiguous 16B LDS chunks per lane
            const __bf16* arow = sX + (rt * 16 + m) * DIM + k0;
            v8bf alo = *(const v8bf*)(arow + 8 * half);
            v8bf ahi = *(const v8bf*)(arow + 16 + 8 * half);
            v16bf a = __builtin_shufflevector(alo, ahi,
                0, 1, 2, 3, 4, 5, 6, 7, 8, 9, 10, 11, 12, 13, 14, 15);
            // B fragment: two contiguous 16B global chunks per lane (bf16 weights)
            v8bf blo = *(const v8bf*)(brow_base + k0 + 8 * half);
            v8bf bhi = *(const v8bf*)(brow_base + k0 + 16 + 8 * half);
            v16bf b = __builtin_shufflevector(blo, bhi,
                0, 1, 2, 3, 4, 5, 6, 7, 8, 9, 10, 11, 12, 13, 14, 15);
            acc = __builtin_amdgcn_wmma_f32_16x16x32_bf16(
                false, a, false, b, (short)0, acc, false, false);
        }
        const float bb = b1[e * HID + ncol];
#pragma unroll
        for (int r = 0; r < 8; ++r) {
            int mrow = rt * 16 + r + half * 8;     // C layout: vgpr r, lane half -> M
            float hv = gelu_tanh(acc[r] + bb);
            sH[mrow * HID + ncol] = f2bf(hv);
        }
    }
    __syncthreads();

    // ---- phase 2: o = h W2^T + b2; out[t] += gate_w * o ----
    const __bf16* W2e = w2 + (size_t)e * DIM * HID;
    for (int it = 0; it < 8; ++it) {
        const int tt = wave + 8 * it;              // 64 output tiles / 8 waves
        const int rt2 = tt & 1, ct2 = tt >> 1;
        const int ncol = ct2 * 16 + m;
        v8f acc = {};
        const __bf16* brow_base = W2e + (size_t)ncol * HID;
        for (int k0 = 0; k0 < HID; k0 += 32) {
            const __bf16* arow = sH + (rt2 * 16 + m) * HID + k0;
            v8bf alo = *(const v8bf*)(arow + 8 * half);
            v8bf ahi = *(const v8bf*)(arow + 16 + 8 * half);
            v16bf a = __builtin_shufflevector(alo, ahi,
                0, 1, 2, 3, 4, 5, 6, 7, 8, 9, 10, 11, 12, 13, 14, 15);
            v8bf blo = *(const v8bf*)(brow_base + k0 + 8 * half);
            v8bf bhi = *(const v8bf*)(brow_base + k0 + 16 + 8 * half);
            v16bf b = __builtin_shufflevector(blo, bhi,
                0, 1, 2, 3, 4, 5, 6, 7, 8, 9, 10, 11, 12, 13, 14, 15);
            acc = __builtin_amdgcn_wmma_f32_16x16x32_bf16(
                false, a, false, b, (short)0, acc, false, false);
        }
        const float bb = b2[e * DIM + ncol];
#pragma unroll
        for (int r = 0; r < 8; ++r) {
            int mrow = rt2 * 16 + r + half * 8;
            if (mrow < nrows) {
                float val = (acc[r] + bb) * sW[mrow];
                atomicAdd(&out[(size_t)sT[mrow] * DIM + ncol], val);
            }
        }
    }
}

__global__ void finalize_kernel(const WsHeader* __restrict__ hdr, float* __restrict__ out_aux) {
    if (threadIdx.x != 0 || blockIdx.x != 0) return;
    const float Tf = (float)T_TOK;
    float pa = hdr->acc_pa / (Tf * (float)NE);
    float pbsum = 0.0f;
    for (int e = 0; e < NE; ++e) { float pm = hdr->acc_pm[e] / Tf; pbsum += pm * (1.0f - pm); }
    float pb = 1.0f / (float)NE - pbsum / (float)NE;
    float penalty = (pa + pb) * 0.01f;
    float z = 0.001f * hdr->acc_z / Tf;
    float mean = 0.0f;
    for (int e = 0; e < NE; ++e) mean += hdr->acc_imp[e];
    mean /= (float)NE;
    float var = 0.0f;
    for (int e = 0; e < NE; ++e) { float d = hdr->acc_imp[e] - mean; var += d * d; }
    var /= (float)NE;
    float imp_loss = 0.01f * var / (mean * mean);
    float load = 0.0f;
    for (int e = 0; e < NE; ++e) load += (hdr->acc_cnt[e] / Tf) * (hdr->acc_rp[e] / Tf);
    load *= 0.01f * (float)NE;
    *out_aux = penalty + z + imp_loss + load;
}

// ---------------- launch ----------------
extern "C" void kernel_launch(void* const* d_in, const int* in_sizes, int n_in,
                              void* d_out, int out_size, void* d_ws, size_t ws_size,
                              hipStream_t stream) {
    const float* x        = (const float*)d_in[0];
    const float* gate_w   = (const float*)d_in[1];
    const float* noise_w  = (const float*)d_in[2];
    const float* bias     = (const float*)d_in[3];
    const float* c_fc_w   = (const float*)d_in[4];
    const float* c_fc_b   = (const float*)d_in[5];
    const float* c_proj_w = (const float*)d_in[6];
    const float* c_proj_b = (const float*)d_in[7];
    const float* noise    = (const float*)d_in[8];
    float* out = (float*)d_out;

    // workspace layout: header | lists | wlist | W1 bf16 | W2 bf16
    char* ws = (char*)d_ws;
    WsHeader* hdr = (WsHeader*)ws;
    int*    lists = (int*)(ws + 256);
    float*  wlist = (float*)(ws + 256 + (size_t)NE * T_TOK * sizeof(int));
    __bf16* w1bf  = (__bf16*)(ws + 256 + (size_t)NE * T_TOK * 8);
    __bf16* w2bf  = w1bf + (size_t)NE * HID * DIM;

    const int nW8 = NE * HID * DIM / 8;   // elements/8 per weight tensor

    // one-time weight down-conversion to bf16 (reads 67MB once; L2-resident after)
    cvt_bf16_kernel<<<(nW8 + 255) / 256, 256, 0, stream>>>(c_fc_w,   w1bf, nW8);
    cvt_bf16_kernel<<<(nW8 + 255) / 256, 256, 0, stream>>>(c_proj_w, w2bf, nW8);

    // zero output (results + aux) and workspace header
    const int n_out = T_TOK * DIM + 1;
    zero_f32<<<(n_out + 255) / 256, 256, 0, stream>>>(out, n_out);
    zero_f32<<<1, 64, 0, stream>>>((float*)ws, 64);

    // router + aux-loss partials + expert token lists
    router_kernel<<<T_TOK / 8, 256, 0, stream>>>(x, gate_w, noise_w, bias, noise,
                                                 hdr, lists, wlist);

    // fused expert FFN (bf16 WMMA), ~160KB LDS per block
    size_t smem = (size_t)TM * DIM * sizeof(__bf16)
                + (size_t)TM * HID * sizeof(__bf16)
                + (size_t)TM * (sizeof(float) + sizeof(int));
    dim3 grid(T_TOK / TM, NE);
    ffn_kernel<<<grid, 256, smem, stream>>>(x, w1bf, c_fc_b, w2bf, c_proj_b,
                                            (int*)ws, lists, wlist, out);

    // combine aux losses -> out[T*D]
    finalize_kernel<<<1, 32, 0, stream>>>(hdr, out + (size_t)T_TOK * DIM);
}
